// GraphClassifierWithMetadata_22325240004778
// MI455X (gfx1250) — matrix operand order repaired and checked by reading, not compiled, not run on hardware
//
#include <hip/hip_runtime.h>
#include <math.h>

// ---------------------------------------------------------------------------
// GraphClassifierWithMetadata for MI455X (gfx1250, wave32)
//
// Roofline: workload is HBM/L2 bound (edge_attr stream ~102MB; h tables are
// 12.8MB -> L2 resident so edge gather/scatter stays on-chip). The two feature
// GEMMs (50000x128x64, 50000x64x64) run on v_wmma_f32_16x16x32_bf16 with:
//   - A tile (16 rows, contiguous 4KB) staged to LDS via the Tensor Data Mover
//     (tensor_load_to_lds + s_wait_tensorcnt), shared by all 4 waves,
//   - B prepacked column-major bf16 so fragments are 2x b128 loads per lane,
//   - K fully unrolled (template) -> back-to-back WMMA accumulate chains.
// Per-edge "edge GEMM" is algebraically collapsed: (ea@We)@a_e == ea.(We@a_e).
// ---------------------------------------------------------------------------

#define Nn   50000
#define Ee   1600000
#define DIN  128
#define Hh   64
#define DEe  16
#define DMm  32
#define Gg   64
#define DOUT 10

typedef __attribute__((ext_vector_type(16))) __bf16 v16bf;
typedef __attribute__((ext_vector_type(8)))  float  v8f;
typedef __attribute__((ext_vector_type(4)))  unsigned int v4u;
typedef __attribute__((ext_vector_type(8)))  int    v8i;
typedef __attribute__((ext_vector_type(4)))  int    v4i;

struct BF16x16 { uint4 lo, hi; };   // 32B container for a v16bf fragment

__device__ __forceinline__ unsigned short f2bf(float f) {
  unsigned u = __float_as_uint(f);
  unsigned r = u + 0x7FFFu + ((u >> 16) & 1u);   // round-to-nearest-even
  return (unsigned short)(r >> 16);
}

__device__ __forceinline__ void atomicMaxFloat(float* addr, float val) {
  unsigned* ua = (unsigned*)addr;
  unsigned cur = *ua, assumed;
  do {
    assumed = cur;
    if (__uint_as_float(assumed) >= val) break;
    cur = atomicCAS(ua, assumed, __float_as_uint(val));
  } while (cur != assumed);
}

// ---------------- small prep kernels ----------------

// v12[0:16) = We1 @ a_e1 ; v12[16:32) = We2 @ a_e2
__global__ void prevec_kernel(const float* __restrict__ We1, const float* __restrict__ ae1,
                              const float* __restrict__ We2, const float* __restrict__ ae2,
                              float* __restrict__ v12) {
  int t = threadIdx.x;            // 0..31
  int layer = t >> 4, k = t & 15;
  const float* We = layer ? We2 : We1;
  const float* ae = layer ? ae2 : ae1;
  float s = 0.f;
  for (int h = 0; h < Hh; ++h) s += We[k * Hh + h] * ae[h];
  v12[layer * 16 + k] = s;
}

// pack W[K,64] (f32 row-major) -> Bt[64,K] (bf16 column-major of W)
__global__ void packBt_kernel(const float* __restrict__ W, unsigned short* __restrict__ Bt, int K) {
  int t = blockIdx.x * 256 + threadIdx.x;
  if (t >= K * Hh) return;
  int n = t / K, k = t % K;
  Bt[t] = f2bf(W[k * Hh + n]);
}

// row-wise L2 normalize x (128 cols) -> bf16. one wave per node, 4 elems/lane.
__global__ void norm_x_kernel(const float* __restrict__ x, unsigned short* __restrict__ xbf, int n) {
  int node = blockIdx.x * 8 + (threadIdx.x >> 5);
  if (node >= n) return;
  int lane = threadIdx.x & 31;
  const float4 v = ((const float4*)(x + (size_t)node * DIN))[lane];
  float ss = v.x * v.x + v.y * v.y + v.z * v.z + v.w * v.w;
#pragma unroll
  for (int o = 16; o > 0; o >>= 1) ss += __shfl_xor(ss, o, 32);
  float inv = 1.0f / fmaxf(sqrtf(ss), 1e-6f);
  unsigned lo = (unsigned)f2bf(v.x * inv) | ((unsigned)f2bf(v.y * inv) << 16);
  unsigned hi = (unsigned)f2bf(v.z * inv) | ((unsigned)f2bf(v.w * inv) << 16);
  uint2 pk; pk.x = lo; pk.y = hi;
  ((uint2*)(xbf + (size_t)node * DIN))[lane] = pk;
}

// per edge: normalize attr, dot with v1/v2, accumulate per-dst count & sums
__global__ void edge_pre_kernel(const int* __restrict__ dst, const float* __restrict__ ea,
                                const float* __restrict__ v12,
                                float* __restrict__ d1, float* __restrict__ d2,
                                float* __restrict__ cnt, float* __restrict__ s1,
                                float* __restrict__ s2, int E) {
  int e = blockIdx.x * 256 + threadIdx.x;
  if (e >= E) return;
  const float* a = ea + (size_t)e * DEe;
  float ss = 0.f, q1 = 0.f, q2 = 0.f;
#pragma unroll
  for (int k = 0; k < DEe; ++k) {
    float t = a[k];
    ss += t * t; q1 += t * v12[k]; q2 += t * v12[16 + k];
  }
  float inv = 1.0f / fmaxf(sqrtf(ss), 1e-6f);
  q1 *= inv; q2 *= inv;
  d1[e] = q1; d2[e] = q2;
  int d = dst[e];
  atomicAdd(&cnt[d], 1.0f);
  atomicAdd(&s1[d], q1);
  atomicAdd(&s2[d], q2);
}

// ------- WMMA GEMM: C[M,64] = A[M,K](bf16 row-major) @ W (as Bt[64,K] bf16) --
// grid.x = M/16, blockDim = 128 (4 waves -> 4 column tiles of 64).
// A tile (16 x K bf16 = contiguous 32*K bytes) is staged to LDS via TDM.
template <int K>
__global__ void gemm_bf16_wmma(const unsigned short* __restrict__ A,
                               const unsigned short* __restrict__ Bt,
                               float* __restrict__ C, int guard) {
  __shared__ unsigned short sA[16 * K];     // only LDS object -> LDS offset 0
  const int lane = threadIdx.x & 31;
  const int wave = threadIdx.x >> 5;
  const int mt   = blockIdx.x;
  const int half = lane >> 4;
  const int l16  = lane & 15;

  if (guard) sA[0] = 0;  // opaque: keep LDS "written" for the optimizer

#if __has_builtin(__builtin_amdgcn_tensor_load_to_lds)
  if (wave == 0) {
    unsigned long long ga = (unsigned long long)(A + (size_t)mt * 16 * K);
    // D# group0: count=1 | lds_addr=0 | global_addr | type=2 ("image")
    v4u g0 = { 1u, 0u,
               (unsigned)(ga & 0xFFFFFFFFu),
               (unsigned)((ga >> 32) & 0x01FFFFFFu) | 0x80000000u };
    // D# group1: data_size=2B; tensor 16 rows x K elems, stride K; tile 16 x K
    v8i g1 = { (int)0x00010000,
               (int)((unsigned)K << 16),    // tensor_dim0 = K
               (int)(16u << 16),            // tensor_dim1 = 16
               (int)((unsigned)K << 16),    // tile_dim0 = K
               16,                          // tile_dim1 = 16
               K,                           // tensor_dim0_stride = K
               0, 0 };
    v4i z4 = { 0, 0, 0, 0 };
    v8i z8 = { 0, 0, 0, 0, 0, 0, 0, 0 };
    __builtin_amdgcn_tensor_load_to_lds(g0, g1, z4, z4, z8, 0);
    __builtin_amdgcn_s_wait_tensorcnt(0);
  }
  __syncthreads();
#else
  for (int i = threadIdx.x; i < 16 * K / 8; i += 128)
    ((uint4*)sA)[i] = ((const uint4*)(A + (size_t)mt * 16 * K))[i];
  __syncthreads();
#endif

  const unsigned short* arow = sA + l16 * K;                       // LDS row
  const unsigned short* bcol = Bt + (size_t)(wave * 16 + l16) * K; // packed col
  v8f acc = {};
#pragma unroll
  for (int k0 = 0; k0 < K; k0 += 32) {
    // A fragment: two contiguous 16B runs (ISA 16-bit A 16x32 layout)
    BF16x16 af, bf;
    af.lo = *(const uint4*)(arow + k0 + half * 8);
    af.hi = *(const uint4*)(arow + k0 + 16 + half * 8);
    // B fragment: two contiguous 16B runs of one column (same K order as A)
    bf.lo = *(const uint4*)(bcol + k0 + half * 8);
    bf.hi = *(const uint4*)(bcol + k0 + 16 + half * 8);
    v16bf av = __builtin_bit_cast(v16bf, af);
    v16bf bv = __builtin_bit_cast(v16bf, bf);
    acc = __builtin_amdgcn_wmma_f32_16x16x32_bf16(false, av, false, bv, (short)0, acc, false, false);
  }
  const int col = wave * 16 + l16;
#pragma unroll
  for (int r = 0; r < 8; ++r)
    C[((size_t)mt * 16 + half * 8 + r) * Hh + col] = acc[r];
}

// ---------------- GAT attention kernels (layer-parameterized) ----------------

__global__ void node_alpha_kernel(const float* __restrict__ h, const float* __restrict__ a_s,
                                  const float* __restrict__ a_d, const float* __restrict__ ssum,
                                  const float* __restrict__ cnt, float* __restrict__ hs,
                                  float* __restrict__ hd, float* __restrict__ aself,
                                  float* __restrict__ m, int n) {
  int node = blockIdx.x * 8 + (threadIdx.x >> 5);
  if (node >= n) return;
  int lane = threadIdx.x & 31;
  const float* hr = h + (size_t)node * Hh;
  float h0 = hr[lane], h1 = hr[lane + 32];
  float s = h0 * a_s[lane] + h1 * a_s[lane + 32];
  float d = h0 * a_d[lane] + h1 * a_d[lane + 32];
#pragma unroll
  for (int o = 16; o > 0; o >>= 1) { s += __shfl_xor(s, o, 32); d += __shfl_xor(d, o, 32); }
  if (lane == 0) {
    float self = s + d + ssum[node] / fmaxf(cnt[node], 1.0f);
    self = self > 0.f ? self : 0.2f * self;            // leaky_relu
    hs[node] = s; hd[node] = d; aself[node] = self; m[node] = self;
  }
}

__global__ void edge_alpha_kernel(const int* __restrict__ src, const int* __restrict__ dst,
                                  const float* __restrict__ hs, const float* __restrict__ hd,
                                  const float* __restrict__ de, float* __restrict__ alpha,
                                  float* __restrict__ m, int E) {
  int e = blockIdx.x * 256 + threadIdx.x;
  if (e >= E) return;
  float a = hs[src[e]] + hd[dst[e]] + de[e];
  a = a > 0.f ? a : 0.2f * a;
  alpha[e] = a;
  atomicMaxFloat(&m[dst[e]], a);
}

__global__ void node_z_kernel(const float* __restrict__ aself, const float* __restrict__ m,
                              float* __restrict__ es, float* __restrict__ z, int n) {
  int i = blockIdx.x * 256 + threadIdx.x;
  if (i >= n) return;
  float e = expf(aself[i] - m[i]);
  es[i] = e; z[i] = e;
}

__global__ void edge_z_kernel(const int* __restrict__ dst, const float* __restrict__ alpha,
                              const float* __restrict__ m, float* __restrict__ z, int E) {
  int e = blockIdx.x * 256 + threadIdx.x;
  if (e >= E) return;
  int d = dst[e];
  atomicAdd(&z[d], expf(alpha[e] - m[d]));
}

// wave per edge: out[dst] += w * h[src], 2 channels per lane
__global__ void edge_agg_kernel(const int* __restrict__ src, const int* __restrict__ dst,
                                const float* __restrict__ alpha, const float* __restrict__ m,
                                const float* __restrict__ z, const float* __restrict__ h,
                                float* __restrict__ out, int E) {
  int e = blockIdx.x * 8 + (threadIdx.x >> 5);
  if (e >= E) return;
  int lane = threadIdx.x & 31;
  int s = src[e], d = dst[e];
  float w = expf(alpha[e] - m[d]) / z[d];
  atomicAdd(&out[(size_t)d * Hh + lane],      w * h[(size_t)s * Hh + lane]);
  atomicAdd(&out[(size_t)d * Hh + lane + 32], w * h[(size_t)s * Hh + lane + 32]);
}

// layer-1 finish: relu(out + w_self*h + b) -> bf16 for next GEMM
__global__ void finish1_kernel(const float* __restrict__ out, const float* __restrict__ h,
                               const float* __restrict__ es, const float* __restrict__ z,
                               const float* __restrict__ b, unsigned short* __restrict__ x1bf, int n) {
  int node = blockIdx.x * 8 + (threadIdx.x >> 5);
  if (node >= n) return;
  int lane = threadIdx.x & 31;
  float ws = es[node] / z[node];
#pragma unroll
  for (int t = 0; t < 2; ++t) {
    int c = lane + 32 * t;
    float v = out[(size_t)node * Hh + c] + ws * h[(size_t)node * Hh + c] + b[c];
    x1bf[(size_t)node * Hh + c] = f2bf(fmaxf(v, 0.f));
  }
}

// layer-2 finish fused with global mean pool accumulation
__global__ void finish_pool_kernel(const float* __restrict__ out, const float* __restrict__ h,
                                   const float* __restrict__ es, const float* __restrict__ z,
                                   const float* __restrict__ b, const int* __restrict__ batch,
                                   float* __restrict__ gsum, float* __restrict__ gcnt, int n) {
  int node = blockIdx.x * 8 + (threadIdx.x >> 5);
  if (node >= n) return;
  int lane = threadIdx.x & 31;
  int g = batch[node];
  float ws = es[node] / z[node];
#pragma unroll
  for (int t = 0; t < 2; ++t) {
    int c = lane + 32 * t;
    float v = fmaxf(out[(size_t)node * Hh + c] + ws * h[(size_t)node * Hh + c] + b[c], 0.f);
    atomicAdd(&gsum[g * Hh + c], v);
  }
  if (lane == 0) atomicAdd(&gcnt[g], 1.0f);
}

// ---------------- MLP head ----------------
__global__ void head1_kernel(const float* __restrict__ gsum, const float* __restrict__ gcnt,
                             const float* __restrict__ meta, const float* __restrict__ fcw1,
                             const float* __restrict__ fcb1, float* __restrict__ hrelu) {
  int g = blockIdx.x, j = threadIdx.x;               // 64 x 64
  float invc = 1.0f / fmaxf(gcnt[g], 1.0f);
  float s = fcb1[j];
  for (int k = 0; k < Hh; ++k)  s += gsum[g * Hh + k] * invc * fcw1[k * Hh + j];
  for (int k = 0; k < DMm; ++k) s += meta[g * DMm + k] * fcw1[(Hh + k) * Hh + j];
  hrelu[g * Hh + j] = fmaxf(s, 0.f);
}

__global__ void head2_kernel(const float* __restrict__ hrelu, const float* __restrict__ fcw2,
                             const float* __restrict__ fcb2, float* __restrict__ outp) {
  int t = blockIdx.x * 64 + threadIdx.x;
  if (t >= Gg * DOUT) return;
  int g = t / DOUT, o = t % DOUT;
  float s = fcb2[o];
  for (int j = 0; j < Hh; ++j) s += hrelu[g * Hh + j] * fcw2[j * DOUT + o];
  outp[t] = s;
}

// ---------------- launch ----------------
extern "C" void kernel_launch(void* const* d_in, const int* in_sizes, int n_in,
                              void* d_out, int out_size, void* d_ws, size_t ws_size,
                              hipStream_t stream) {
  const float* x     = (const float*)d_in[0];
  const int*   ei    = (const int*)d_in[1];
  const float* ea    = (const float*)d_in[2];
  const int*   batch = (const int*)d_in[3];
  const float* meta  = (const float*)d_in[4];
  const float* W1    = (const float*)d_in[5];
  const float* as1   = (const float*)d_in[6];
  const float* ad1   = (const float*)d_in[7];
  const float* We1   = (const float*)d_in[8];
  const float* ae1   = (const float*)d_in[9];
  const float* b1    = (const float*)d_in[10];
  const float* W2    = (const float*)d_in[11];
  const float* as2   = (const float*)d_in[12];
  const float* ad2   = (const float*)d_in[13];
  const float* We2   = (const float*)d_in[14];
  const float* ae2   = (const float*)d_in[15];
  const float* b2    = (const float*)d_in[16];
  const float* fcw1  = (const float*)d_in[17];
  const float* fcb1  = (const float*)d_in[18];
  const float* fcw2  = (const float*)d_in[19];
  const float* fcb2  = (const float*)d_in[20];
  float* outp = (float*)d_out;
  const int* src = ei;
  const int* dst = ei + Ee;

  char* p = (char*)d_ws;
  auto alloc = [&](size_t bytes) -> void* {
    void* r = (void*)p;
    p += (bytes + 255) & ~(size_t)255;
    return r;
  };
  unsigned short* xnbf = (unsigned short*)alloc((size_t)Nn * DIN * 2);
  unsigned short* w1t  = (unsigned short*)alloc((size_t)DIN * Hh * 2);  // [64,128]
  unsigned short* w2t  = (unsigned short*)alloc((size_t)Hh * Hh * 2);   // [64,64]
  unsigned short* x1bf = (unsigned short*)alloc((size_t)Nn * Hh * 2);
  float* v12   = (float*)alloc(32 * 4);
  float* d1    = (float*)alloc((size_t)Ee * 4);
  float* d2    = (float*)alloc((size_t)Ee * 4);
  float* cnt   = (float*)alloc((size_t)3 * Nn * 4);   // cnt | s1 | s2 contiguous
  float* s1 = cnt + Nn;
  float* s2 = cnt + 2 * Nn;
  float* h     = (float*)alloc((size_t)Nn * Hh * 4);
  float* obuf  = (float*)alloc((size_t)Nn * Hh * 4);
  float* alpha = (float*)alloc((size_t)Ee * 4);
  float* hs    = (float*)alloc((size_t)Nn * 4);
  float* hd    = (float*)alloc((size_t)Nn * 4);
  float* aself = (float*)alloc((size_t)Nn * 4);
  float* m     = (float*)alloc((size_t)Nn * 4);
  float* z     = (float*)alloc((size_t)Nn * 4);
  float* es    = (float*)alloc((size_t)Nn * 4);
  float* gsum  = (float*)alloc((size_t)(Gg * Hh + Gg) * 4);  // gsum | gcnt
  float* gcnt  = gsum + Gg * Hh;
  float* hrelu = (float*)alloc((size_t)Gg * Hh * 4);

  const int NW = (Nn + 7) / 8;        // wave-per-node grids
  const int ET = (Ee + 255) / 256;    // thread-per-edge grids
  const int EW = (Ee + 7) / 8;        // wave-per-edge grids

  (void)hipMemsetAsync(cnt, 0, (size_t)3 * Nn * 4, stream);
  (void)hipMemsetAsync(gsum, 0, (size_t)(Gg * Hh + Gg) * 4, stream);

  prevec_kernel<<<1, 32, 0, stream>>>(We1, ae1, We2, ae2, v12);
  packBt_kernel<<<(DIN * Hh + 255) / 256, 256, 0, stream>>>(W1, w1t, DIN);
  packBt_kernel<<<(Hh * Hh + 255) / 256, 256, 0, stream>>>(W2, w2t, Hh);
  norm_x_kernel<<<NW, 256, 0, stream>>>(x, xnbf, Nn);
  edge_pre_kernel<<<ET, 256, 0, stream>>>(dst, ea, v12, d1, d2, cnt, s1, s2, Ee);

  // ---- GAT layer 1 ----
  gemm_bf16_wmma<DIN><<<Nn / 16, 128, 0, stream>>>(xnbf, w1t, h, 0);
  node_alpha_kernel<<<NW, 256, 0, stream>>>(h, as1, ad1, s1, cnt, hs, hd, aself, m, Nn);
  edge_alpha_kernel<<<ET, 256, 0, stream>>>(src, dst, hs, hd, d1, alpha, m, Ee);
  node_z_kernel<<<(Nn + 255) / 256, 256, 0, stream>>>(aself, m, es, z, Nn);
  edge_z_kernel<<<ET, 256, 0, stream>>>(dst, alpha, m, z, Ee);
  (void)hipMemsetAsync(obuf, 0, (size_t)Nn * Hh * 4, stream);
  edge_agg_kernel<<<EW, 256, 0, stream>>>(src, dst, alpha, m, z, h, obuf, Ee);
  finish1_kernel<<<NW, 256, 0, stream>>>(obuf, h, es, z, b1, x1bf, Nn);

  // ---- GAT layer 2 ----
  gemm_bf16_wmma<Hh><<<Nn / 16, 128, 0, stream>>>(x1bf, w2t, h, 0);
  node_alpha_kernel<<<NW, 256, 0, stream>>>(h, as2, ad2, s2, cnt, hs, hd, aself, m, Nn);
  edge_alpha_kernel<<<ET, 256, 0, stream>>>(src, dst, hs, hd, d2, alpha, m, Ee);
  node_z_kernel<<<(Nn + 255) / 256, 256, 0, stream>>>(aself, m, es, z, Nn);
  edge_z_kernel<<<ET, 256, 0, stream>>>(dst, alpha, m, z, Ee);
  (void)hipMemsetAsync(obuf, 0, (size_t)Nn * Hh * 4, stream);
  edge_agg_kernel<<<EW, 256, 0, stream>>>(src, dst, alpha, m, z, h, obuf, Ee);
  finish_pool_kernel<<<NW, 256, 0, stream>>>(obuf, h, es, z, b2, batch, gsum, gcnt, Nn);

  // ---- head ----
  head1_kernel<<<Gg, Hh, 0, stream>>>(gsum, gcnt, meta, fcw1, fcb1, hrelu);
  head2_kernel<<<(Gg * DOUT + 63) / 64, 64, 0, stream>>>(hrelu, fcw2, fcb2, outp);
}
